// DualAttention_44341242364496
// MI455X (gfx1250) — matrix operand order repaired and checked by reading, not compiled, not run on hardware
//
#include <hip/hip_runtime.h>

#define BB 4
#define NN 4096
#define CC 256
#define CQ 32          // C/8
#define TOK 8          // tokens per projection block

typedef __attribute__((ext_vector_type(16))) __bf16 v16bf;
typedef __attribute__((ext_vector_type(8)))  float  v8f;
typedef __attribute__((ext_vector_type(4)))  int    i4;

union Frag { v16bf bf; i4 q[2]; };

// A-matrix 16x32 bf16 fragment (ISA layout: half0 K0-7 & K16-23, half1 K8-15 & K24-31).
// p points at element [row0][0] of the tile; stride in elements.
__device__ __forceinline__ v16bf load_a(const __bf16* p, int stride, int kbase) {
    const int l = threadIdx.x & 31;
    const __bf16* row = p + (size_t)(l & 15) * stride + kbase + ((l & 16) ? 8 : 0);
    Frag f;
    f.q[0] = *(const i4*)(row);
    f.q[1] = *(const i4*)(row + 16);
    return f.bf;
}

// B-matrix 32x16 bf16 fragment (half0 K0-7 & K16-23, half1 K8-15 & K24-31); lane selects column,
// columns are stored as rows of a transposed buffer (stride elements apart).
__device__ __forceinline__ v16bf load_b(const __bf16* p, int stride, int kbase) {
    const int l = threadIdx.x & 31;
    const __bf16* row = p + (size_t)(l & 15) * stride + kbase + ((l & 16) ? 8 : 0);
    Frag f;
    f.q[0] = *(const i4*)(row);
    f.q[1] = *(const i4*)(row + 16);
    return f.bf;
}

__device__ __forceinline__ v8f wmma_bf16(v16bf a, v16bf b, v8f c) {
    return __builtin_amdgcn_wmma_f32_16x16x32_bf16(false, a, false, b, (short)0, c,
                                                   false, false);
}

// async DMA of one 32x32 bf16 K tile (2KB) into LDS; issued by one wave,
// lane = key row, 4x b128 per lane (offset applies to both LDS and global side).
__device__ __forceinline__ void async_copy_ktile(uint32_t lds_addr, uint64_t gaddr) {
    asm volatile(
        "global_load_async_to_lds_b128 %0, %1, off\n\t"
        "global_load_async_to_lds_b128 %0, %1, off offset:16\n\t"
        "global_load_async_to_lds_b128 %0, %1, off offset:32\n\t"
        "global_load_async_to_lds_b128 %0, %1, off offset:48"
        :: "v"(lds_addr), "v"(gaddr) : "memory");
}

// reductions across a 16-lane half (matches C/D layout: cols 0-15 live in one half)
__device__ __forceinline__ float redmax16(float v) {
    v = fmaxf(v, __shfl_xor(v, 1, 32));
    v = fmaxf(v, __shfl_xor(v, 2, 32));
    v = fmaxf(v, __shfl_xor(v, 4, 32));
    v = fmaxf(v, __shfl_xor(v, 8, 32));
    return v;
}
__device__ __forceinline__ float redsum16(float v) {
    v += __shfl_xor(v, 1, 32);
    v += __shfl_xor(v, 2, 32);
    v += __shfl_xor(v, 4, 32);
    v += __shfl_xor(v, 8, 32);
    return v;
}
__device__ __forceinline__ float redmax32(float v) {
    v = redmax16(v);
    v = fmaxf(v, __shfl_xor(v, 16, 32));
    return v;
}
__device__ __forceinline__ float redsum32(float v) {
    v = redsum16(v);
    v += __shfl_xor(v, 16, 32);
    return v;
}

// -------- kernel 1: projections + bf16 staging ------------------------------
__global__ void proj_kernel(const float* __restrict__ pts,
                            const float* __restrict__ Wq, const float* __restrict__ bq,
                            const float* __restrict__ Wk, const float* __restrict__ bk,
                            const float* __restrict__ Wv, const float* __restrict__ bv,
                            __bf16* __restrict__ qbf, __bf16* __restrict__ kbf,
                            __bf16* __restrict__ vT,  __bf16* __restrict__ pRow,
                            __bf16* __restrict__ pT) {
    __shared__ float sp[TOK][CC];
    const int b  = blockIdx.x / (NN / TOK);
    const int n0 = (blockIdx.x % (NN / TOK)) * TOK;
    const int t  = threadIdx.x;   // 0..255
    const float* prow = pts + ((size_t)b * NN + n0) * CC;
    for (int j = 0; j < TOK; ++j) sp[j][t] = prow[(size_t)j * CC + t];
    __syncthreads();

    for (int j = 0; j < TOK; ++j) {
        float x = sp[j][t];
        pRow[((size_t)b * NN + n0 + j) * CC + t] = (__bf16)x;
        pT[((size_t)b * CC + t) * NN + n0 + j]   = (__bf16)x;
    }
    // v projection: thread t owns output channel t, reuses Wv row across 8 tokens
    float av[TOK];
#pragma unroll
    for (int j = 0; j < TOK; ++j) av[j] = bv[t];
    const float* wv = Wv + (size_t)t * CC;
    for (int i = 0; i < CC; ++i) {
        float w = wv[i];
#pragma unroll
        for (int j = 0; j < TOK; ++j) av[j] = fmaf(w, sp[j][i], av[j]);
    }
    for (int j = 0; j < TOK; ++j)
        vT[((size_t)b * CC + t) * NN + n0 + j] = (__bf16)av[j];

    if (t < 2 * CQ) {                      // q (threads 0-31) and k (32-63)
        const int  qt  = t & 31;
        const bool isK = t >= CQ;
        const float* wr = (isK ? Wk : Wq) + (size_t)qt * CC;
        float bias = isK ? bk[qt] : bq[qt];
        float a[TOK];
#pragma unroll
        for (int j = 0; j < TOK; ++j) a[j] = bias;
        for (int i = 0; i < CC; ++i) {
            float w = wr[i];
#pragma unroll
            for (int j = 0; j < TOK; ++j) a[j] = fmaf(w, sp[j][i], a[j]);
        }
        __bf16* dst = isK ? kbf : qbf;
        for (int j = 0; j < TOK; ++j)
            dst[((size_t)b * NN + n0 + j) * CQ + qt] = (__bf16)a[j];
    }
}

// -------- kernel 2: channel energy = points^T @ points (256x256, inner 4096) --
__global__ void chan_energy_kernel(const __bf16* __restrict__ pT,
                                   float* __restrict__ energy) {
    const int wave  = threadIdx.x >> 5;
    const int gtile = blockIdx.x * 8 + wave;      // 0 .. B*256-1
    const int b  = gtile >> 8;
    const int tt = gtile & 255;
    const int ti = tt >> 4, tj = tt & 15;
    const __bf16* base = pT + (size_t)b * CC * NN;
    v8f acc = {};
    for (int k = 0; k < NN; k += 32) {
        v16bf a  = load_a(base + (size_t)(ti * 16) * NN, NN, k);
        v16bf bb = load_b(base + (size_t)(tj * 16) * NN, NN, k);
        acc = wmma_bf16(a, bb, acc);
    }
    const int l    = threadIdx.x & 31;
    const int row0 = ti * 16 + ((l & 16) ? 8 : 0);
    const int col  = tj * 16 + (l & 15);
    float* eb = energy + (size_t)b * CC * CC;
#pragma unroll
    for (int r = 0; r < 8; ++r)
        eb[(size_t)(row0 + r) * CC + col] = acc[r];
}

// -------- kernel 3: softmax of (rowmax - energy) over last dim ---------------
__global__ void chan_softmax_kernel(const float* __restrict__ energy,
                                    __bf16* __restrict__ attnc) {
    const int row = blockIdx.x * 8 + (threadIdx.x >> 5);  // 0..B*C-1
    const int l   = threadIdx.x & 31;
    const float* e = energy + (size_t)row * CC;
    float v[8], m1 = -1e30f;
#pragma unroll
    for (int i = 0; i < 8; ++i) { v[i] = e[l + 32 * i]; m1 = fmaxf(m1, v[i]); }
    m1 = redmax32(m1);
    float z[8], zmax = -1e30f;
#pragma unroll
    for (int i = 0; i < 8; ++i) { z[i] = m1 - v[i]; zmax = fmaxf(zmax, z[i]); }
    zmax = redmax32(zmax);
    float p[8], s = 0.f;
#pragma unroll
    for (int i = 0; i < 8; ++i) { p[i] = __expf(z[i] - zmax); s += p[i]; }
    s = redsum32(s);
    float inv = 1.0f / s;
    __bf16* a = attnc + (size_t)row * CC;
#pragma unroll
    for (int i = 0; i < 8; ++i) a[l + 32 * i] = (__bf16)(p[i] * inv);
}

// -------- kernel 4: fused flash point-attention + channel GEMM + combine -----
__global__ void attn_out_kernel(const float* __restrict__ pts,
                                const __bf16* __restrict__ qbf,
                                const __bf16* __restrict__ kbf,
                                const __bf16* __restrict__ vT,
                                const __bf16* __restrict__ pRow,
                                const __bf16* __restrict__ attnc,
                                const float* __restrict__ gamma_p,
                                float* __restrict__ out) {
    // double-buffered K tile (32 keys x 32 dims bf16 = 2KB each), DMA'd by wave 0
    __shared__ __attribute__((aligned(16))) __bf16 kbuf[2][32 * CQ];
    __shared__ __attribute__((aligned(16))) __bf16 pbuf[8][16 * 32];
    const int wave = threadIdx.x >> 5;
    const int l    = threadIdx.x & 31;
    const int b  = blockIdx.x / (NN / 16);
    const int n0 = (blockIdx.x % (NN / 16)) * 16;
    const int cw = wave * 32;               // this wave's 32 output channels
    const float g = gamma_p[0];

    const __bf16* qb  = qbf + ((size_t)b * NN + n0) * CQ;
    const __bf16* kb  = kbf + (size_t)b * NN * CQ;
    const __bf16* vtb = vT  + (size_t)b * CC * NN;

    // flat->LDS: hardware uses addr[31:0] as the LDS byte address
    const uint32_t klds = (uint32_t)(size_t)(&kbuf[0][0]) + (uint32_t)l * 64;

    v16bf qfrag = load_a(qb, CQ, 0);        // Q tile, head dim 32 == one K step
    v8f accp0 = {}, accp1 = {};
    float m[8], lsum[8];
#pragma unroll
    for (int r = 0; r < 8; ++r) { m[r] = -1e30f; lsum[r] = 0.f; }
    __bf16* P = &pbuf[wave][0];

    // prologue: stage K tile 0 into kbuf[0]
    if (wave == 0)
        async_copy_ktile(klds, (uint64_t)(size_t)kb + (uint64_t)l * 64);

    for (int kt = 0; kt < NN; kt += 32) {
        const int cur = (kt >> 5) & 1;
        if (kt + 32 < NN) {
            if (wave == 0)                  // prefetch next K tile into other buffer
                async_copy_ktile(klds + (uint32_t)((cur ^ 1) * 2048),
                                 (uint64_t)(size_t)(kb + (size_t)(kt + 32) * CQ) +
                                     (uint64_t)l * 64);
            asm volatile("s_wait_asynccnt 0x4" ::: "memory");  // older 4 (cur) done
        } else {
            asm volatile("s_wait_asynccnt 0x0" ::: "memory");
        }
        __syncthreads();                    // publish kbuf[cur] to all waves

        const __bf16* kc = &kbuf[cur][0];
        v16bf k0 = load_b(kc, 32, 0);
        v16bf k1 = load_b(kc + 16 * 32, 32, 0);
        v8f s0 = {}; s0 = wmma_bf16(qfrag, k0, s0);   // S[:,kt..kt+15]
        v8f s1 = {}; s1 = wmma_bf16(qfrag, k1, s1);   // S[:,kt+16..kt+31]
#pragma unroll
        for (int r = 0; r < 8; ++r) {
            float mx = redmax16(fmaxf(s0[r], s1[r]));
            float mnew  = fmaxf(m[r], mx);
            float alpha = __expf(m[r] - mnew);
            float e0 = __expf(s0[r] - mnew);
            float e1 = __expf(s1[r] - mnew);
            lsum[r] = lsum[r] * alpha + redsum16(e0 + e1);
            m[r] = mnew;
            accp0[r] *= alpha;
            accp1[r] *= alpha;
            const int prow = r + ((l & 16) ? 8 : 0);
            P[prow * 32 + (l & 15)]      = (__bf16)e0;
            P[prow * 32 + 16 + (l & 15)] = (__bf16)e1;
        }
        __syncthreads();                      // order LDS P stores vs A-frag loads
        v16bf pf = load_a(P, 32, 0);          // P as 16x32 A fragment
        v16bf v0 = load_b(vtb + (size_t)cw * NN,        NN, kt);
        v16bf v1 = load_b(vtb + (size_t)(cw + 16) * NN, NN, kt);
        accp0 = wmma_bf16(pf, v0, accp0);
        accp1 = wmma_bf16(pf, v1, accp1);
    }
#pragma unroll
    for (int r = 0; r < 8; ++r) {
        float inv = 1.0f / lsum[r];
        accp0[r] *= inv;
        accp1[r] *= inv;
    }

    // channel branch: out_c tile = points(n0..n0+15,:) @ attn_c^T (cols cw..cw+31)
    v8f accc0 = {}, accc1 = {};
    const __bf16* prb = pRow  + ((size_t)b * NN + n0) * CC;
    const __bf16* acb = attnc + (size_t)b * CC * CC;
    for (int kd = 0; kd < CC; kd += 32) {
        v16bf a  = load_a(prb, CC, kd);
        v16bf b0 = load_b(acb + (size_t)cw * CC,        CC, kd);
        v16bf b1 = load_b(acb + (size_t)(cw + 16) * CC, CC, kd);
        accc0 = wmma_bf16(a, b0, accc0);
        accc1 = wmma_bf16(a, b1, accc1);
    }

    const int row0 = n0 + ((l & 16) ? 8 : 0);
    const int cA = cw + (l & 15);
    const int cB = cw + 16 + (l & 15);
    const float* pin = pts + (size_t)b * NN * CC;
    float* ob = out + (size_t)b * NN * CC;
#pragma unroll
    for (int r = 0; r < 8; ++r) {
        int row = row0 + r;
        float pA = pin[(size_t)row * CC + cA];
        float pB = pin[(size_t)row * CC + cB];
        ob[(size_t)row * CC + cA] = g * (accp0[r] + accc0[r]) + 2.f * pA;
        ob[(size_t)row * CC + cB] = g * (accp1[r] + accc1[r]) + 2.f * pB;
    }
}

extern "C" void kernel_launch(void* const* d_in, const int* in_sizes, int n_in,
                              void* d_out, int out_size, void* d_ws, size_t ws_size,
                              hipStream_t stream) {
    const float* pts = (const float*)d_in[0];
    const float* Wq  = (const float*)d_in[1];
    const float* bq  = (const float*)d_in[2];
    const float* Wk  = (const float*)d_in[3];
    const float* bk  = (const float*)d_in[4];
    const float* Wv  = (const float*)d_in[5];
    const float* bv  = (const float*)d_in[6];
    const float* gam = (const float*)d_in[7];
    float* out = (float*)d_out;

    char* ws = (char*)d_ws;
    size_t off = 0;
    auto alloc = [&](size_t bytes) -> char* {
        char* p = ws + off;
        off = (off + bytes + 255) & ~(size_t)255;
        return p;
    };
    __bf16* qbf   = (__bf16*)alloc((size_t)BB * NN * CQ * 2);
    __bf16* kbf   = (__bf16*)alloc((size_t)BB * NN * CQ * 2);
    __bf16* vT    = (__bf16*)alloc((size_t)BB * CC * NN * 2);
    __bf16* pRow  = (__bf16*)alloc((size_t)BB * NN * CC * 2);
    __bf16* pT    = (__bf16*)alloc((size_t)BB * CC * NN * 2);
    float*  ener  = (float*) alloc((size_t)BB * CC * CC * 4);
    __bf16* attnc = (__bf16*)alloc((size_t)BB * CC * CC * 2);

    proj_kernel<<<BB * NN / TOK, 256, 0, stream>>>(pts, Wq, bq, Wk, bk, Wv, bv,
                                                   qbf, kbf, vT, pRow, pT);
    chan_energy_kernel<<<BB * 256 / 8, 256, 0, stream>>>(pT, ener);
    chan_softmax_kernel<<<BB * CC / 8, 256, 0, stream>>>(ener, attnc);
    attn_out_kernel<<<BB * (NN / 16), 256, 0, stream>>>(pts, qbf, kbf, vT, pRow,
                                                        attnc, gam, out);
}
